// VoiceCloningModel_16054587752846
// MI455X (gfx1250) — matrix operand order repaired and checked by reading, not compiled, not run on hardware
//
#include <hip/hip_runtime.h>

// ---------------------------------------------------------------------------
// Fused 4-layer MLP for MI455X (gfx1250, wave32, WMMA).
//   h1 = relu(x @ W1 + b1); h2 = relu(h1 @ W2 + b2);
//   h3 = relu(h2 @ W3 + b3); out = h3 @ W4 + b4
// v_wmma_f32_16x16x32_bf16 everywhere; weights prepacked into B-fragment
// layout in d_ws; activations ping-pong in LDS (never touch HBM).
// Waves own N-tiles: each B fragment is loaded once and reused across the
// block's 8 row-strips (8 independent back-to-back WMMAs per fragment).
// 8 accumulators per wave (64 VGPRs) -- deliberately sized to avoid spills
// (the 16-accumulator variant spilled to scratch).
// ---------------------------------------------------------------------------

typedef __attribute__((ext_vector_type(16))) __bf16 v16bf;
typedef __attribute__((ext_vector_type(8)))  __bf16 v8bf;
typedef __attribute__((ext_vector_type(8)))  float  v8f;

#define IN_DIM   13
#define HID      256
#define OUT_DIM  80
#define TILE_M   128     // rows per block: 8 strips x 16 rows
#define NSTRIPS  8
#define HS       264     // LDS row stride (bf16) for hidden acts: 256 + 8 pad
#define XS       40      // LDS row stride (bf16) for staged x: 32 + 8 pad

// Packed weight fragments: 32(K) x 16(N) bf16, lane-major [frag][lane][16].
// B lane layout (16-bit, 32x16): lane n<16 -> col n, K{0..7}U{16..23};
// lane n>=16 -> col n-16, K{8..15}U{24..31} (K pairs packed per dword).
#define FRAG_ELEMS 512
#define W1_FRAGS  (16 * 1)
#define W2_FRAGS  (16 * 8)
#define W3_FRAGS  (16 * 8)
#define W4_FRAGS  (5  * 8)
#define W1_OFF    0
#define W2_OFF    (W1_OFF + W1_FRAGS * FRAG_ELEMS)
#define W3_OFF    (W2_OFF + W2_FRAGS * FRAG_ELEMS)
#define W4_OFF    (W3_OFF + W3_FRAGS * FRAG_ELEMS)
#define WTOTAL    (W4_OFF + W4_FRAGS * FRAG_ELEMS)   // 159744 elems = 312 KB

// ---------------------------------------------------------------------------
// Prepack: f32 row-major [K][N] weights -> bf16 WMMA B-fragments in d_ws.
// ---------------------------------------------------------------------------
__global__ void prepack_weights(const float* __restrict__ W1,
                                const float* __restrict__ W2,
                                const float* __restrict__ W3,
                                const float* __restrict__ W4,
                                __bf16* __restrict__ wf) {
  int idx = blockIdx.x * blockDim.x + threadIdx.x;
  if (idx >= WTOTAL) return;

  const float* W; int off, ktiles, Ksrc, ldw;
  if (idx < W2_OFF)      { W = W1; off = idx - W1_OFF; ktiles = 1; Ksrc = IN_DIM; ldw = HID; }
  else if (idx < W3_OFF) { W = W2; off = idx - W2_OFF; ktiles = 8; Ksrc = HID;    ldw = HID; }
  else if (idx < W4_OFF) { W = W3; off = idx - W3_OFF; ktiles = 8; Ksrc = HID;    ldw = HID; }
  else                   { W = W4; off = idx - W4_OFF; ktiles = 8; Ksrc = HID;    ldw = OUT_DIM; }

  int frag = off >> 9;
  int lane = (off >> 4) & 31;
  int j    = off & 15;
  int nt   = frag / ktiles;
  int kt   = frag - nt * ktiles;
  int k    = kt * 32 + ((lane & 16) ? 8 : 0) + j + (j & 8);   // B layout K index
  int col  = nt * 16 + (lane & 15);                  // B layout N index
  float v  = (k < Ksrc) ? W[k * ldw + col] : 0.0f;   // zero-pad K (layer 1)
  wf[idx] = (__bf16)v;
}

// ---------------------------------------------------------------------------
// A-fragment load from LDS. 16-bit A 16x32 layout: lane<16 -> M=lane,
// K{0..7}U{16..23}; lane>=16 -> M=lane-16, K{8..15}U{24..31}.
// Two 16 B ds_load_b128 per fragment; +8 stride pad keeps rows in
// disjoint bank groups across lanes.
// ---------------------------------------------------------------------------
__device__ __forceinline__ v16bf load_a(const __bf16* in, int stride,
                                        int row, int kt, int hi) {
  const __bf16* p = in + row * stride + kt * 32 + hi * 8;
  v16bf a;
  ((v8bf*)&a)[0] = *(const v8bf*)(p);
  ((v8bf*)&a)[1] = *(const v8bf*)(p + 16);
  return a;
}

// One Linear(+bias+ReLU) layer, LDS bf16 -> LDS bf16.
// Wave `wv` owns N-tiles {2*wv, 2*wv+1}; 8 accumulators cover all row strips.
__device__ __forceinline__ void layer_bf16(const __bf16* in, int instride,
                                           __bf16* outp, int outstride,
                                           const __bf16* __restrict__ wf,
                                           const float* __restrict__ bias,
                                           int ktiles, int wv, int lane) {
  const int m  = lane & 15;
  const int hi = (lane >> 4) & 1;

  for (int ni = 0; ni < 2; ++ni) {
    const int nt = wv * 2 + ni;
    v8f acc[NSTRIPS] = {};
    const __bf16* wn = wf + (size_t)(nt * ktiles) * FRAG_ELEMS + lane * 16;
    for (int kt = 0; kt < ktiles; ++kt) {
      v16bf b = *(const v16bf*)(wn + kt * FRAG_ELEMS);   // 1 global frag load
#pragma unroll
      for (int rs = 0; rs < NSTRIPS; ++rs) {             // reused 8x
        v16bf a = load_a(in, instride, rs * 16 + m, kt, hi);
        acc[rs] = __builtin_amdgcn_wmma_f32_16x16x32_bf16(
            false, a, false, b, (short)0, acc[rs], false, false);
      }
    }
    const int col = nt * 16 + m;
    const float bv = bias[col];
#pragma unroll
    for (int rs = 0; rs < NSTRIPS; ++rs) {
      const int rbase = rs * 16 + hi * 8;
#pragma unroll
      for (int r = 0; r < 8; ++r) {
        float v = acc[rs][r] + bv;
        v = v > 0.0f ? v : 0.0f;                         // ReLU
        outp[(rbase + r) * outstride + col] = (__bf16)v;
      }
    }
  }
}

// ---------------------------------------------------------------------------
// Fused MLP kernel: 256 threads = 8 waves; block tile = 128 rows.
// ---------------------------------------------------------------------------
__global__ void mlp4_fused(const float* __restrict__ x,
                           const float* __restrict__ b1,
                           const float* __restrict__ b2,
                           const float* __restrict__ b3,
                           const float* __restrict__ b4,
                           const __bf16* __restrict__ wf,
                           float* __restrict__ out) {
  extern __shared__ __bf16 smem[];          // 2 x [TILE_M][HS] bf16 = 132 KB
  __bf16* h0 = smem;
  __bf16* h1 = smem + TILE_M * HS;          // front reused as x stage [TILE_M][XS]

  const int tid  = threadIdx.x;
  const int lane = tid & 31;
  const int wv   = tid >> 5;
  const long row0 = (long)blockIdx.x * TILE_M;

  // Stage x tile: f32 -> bf16, K padded 13 -> 32 with zeros.
  for (int i = tid; i < TILE_M * 32; i += blockDim.x) {
    int r = i >> 5, c = i & 31;
    float v = (c < IN_DIM) ? x[(row0 + r) * IN_DIM + c] : 0.0f;
    h1[r * XS + c] = (__bf16)v;
  }
  __syncthreads();

  layer_bf16(h1, XS, h0, HS, wf + W1_OFF, b1, /*ktiles=*/1, wv, lane);
  __syncthreads();
  layer_bf16(h0, HS, h1, HS, wf + W2_OFF, b2, /*ktiles=*/8, wv, lane);
  __syncthreads();
  layer_bf16(h1, HS, h0, HS, wf + W3_OFF, b3, /*ktiles=*/8, wv, lane);
  __syncthreads();

  // Final layer: N=80 -> 5 N-tiles, one per wave 0..4; no ReLU; f32 -> HBM.
  if (wv < 5) {
    const int m  = lane & 15;
    const int hi = (lane >> 4) & 1;
    const int nt = wv;
    v8f acc[NSTRIPS] = {};
    const __bf16* wn = wf + W4_OFF + (size_t)(nt * 8) * FRAG_ELEMS + lane * 16;
    for (int kt = 0; kt < 8; ++kt) {
      v16bf b = *(const v16bf*)(wn + kt * FRAG_ELEMS);
#pragma unroll
      for (int rs = 0; rs < NSTRIPS; ++rs) {
        v16bf a = load_a(h0, HS, rs * 16 + m, kt, hi);
        acc[rs] = __builtin_amdgcn_wmma_f32_16x16x32_bf16(
            false, a, false, b, (short)0, acc[rs], false, false);
      }
    }
    const int col = nt * 16 + m;
    const float bv = b4[col];
#pragma unroll
    for (int rs = 0; rs < NSTRIPS; ++rs) {
      const int rbase = rs * 16 + hi * 8;
#pragma unroll
      for (int r = 0; r < 8; ++r) {
        out[(row0 + rbase + r) * OUT_DIM + col] = acc[rs][r] + bv;
      }
    }
  }
}

// ---------------------------------------------------------------------------
// Launch. d_in order: x, W1, b1, W2, b2, W3, b3, W4, b4 (all f32).
// ---------------------------------------------------------------------------
extern "C" void kernel_launch(void* const* d_in, const int* in_sizes, int n_in,
                              void* d_out, int out_size, void* d_ws, size_t ws_size,
                              hipStream_t stream) {
  const float* x   = (const float*)d_in[0];
  const float* W1  = (const float*)d_in[1];
  const float* b1v = (const float*)d_in[2];
  const float* W2  = (const float*)d_in[3];
  const float* b2v = (const float*)d_in[4];
  const float* W3  = (const float*)d_in[5];
  const float* b3v = (const float*)d_in[6];
  const float* W4  = (const float*)d_in[7];
  const float* b4v = (const float*)d_in[8];
  __bf16* wf = (__bf16*)d_ws;               // 312 KB of packed bf16 fragments
  float* out = (float*)d_out;

  const int rows = in_sizes[0] / IN_DIM;    // 64 * 2048 = 131072

  prepack_weights<<<(WTOTAL + 255) / 256, 256, 0, stream>>>(W1, W2, W3, W4, wf);

  const int nblocks = rows / TILE_M;        // 1024
  const size_t shbytes = (size_t)2 * TILE_M * HS * sizeof(__bf16);  // 132 KB
  mlp4_fused<<<nblocks, 256, shbytes, stream>>>(x, b1v, b2v, b3v, b4v, wf, out);
}
